// Attention_71966472012100
// MI455X (gfx1250) — matrix operand order repaired and checked by reading, not compiled, not run on hardware
//
#include <hip/hip_runtime.h>
#include <hip/hip_bf16.h>

// ---- problem constants ----
#define Bb   4
#define Sq   2048
#define Hh   12
#define Dd   100
#define HID  1200
#define HIDP 1216     // HID padded to multiple of 32
#define DP   128      // head dim padded to multiple of 32
#define MROWS 8192    // B*S
#define BH   48       // B*H

typedef __attribute__((ext_vector_type(4)))  unsigned int v4u;
typedef __attribute__((ext_vector_type(8)))  unsigned int v8u;
typedef __attribute__((ext_vector_type(16))) __bf16      v16bf;
typedef __attribute__((ext_vector_type(8)))  float       v8f;

// Build a 16-element bf16 WMMA fragment from two 16-byte chunks.
static __device__ inline v16bf frag16(const __bf16* p0, const __bf16* p1) {
  v4u a = *(const v4u*)p0;
  v4u b = *(const v4u*)p1;
  v8u r;
  r[0]=a[0]; r[1]=a[1]; r[2]=a[2]; r[3]=a[3];
  r[4]=b[0]; r[5]=b[1]; r[6]=b[2]; r[7]=b[3];
  return __builtin_bit_cast(v16bf, r);
}

static __device__ inline v8f wmma_bf16(v16bf a, v16bf b, v8f c) {
  return __builtin_amdgcn_wmma_f32_16x16x32_bf16(false, a, false, b, (short)0, c,
                                                 false, false);
}

// ---- f32 -> bf16 with row padding (dst pre-zeroed) ----
__global__ void cvt_pad_kernel(const float* __restrict__ src,
                               __bf16* __restrict__ dst,
                               int rows, int cols, int colsp) {
  long long i = (long long)blockIdx.x * 256 + threadIdx.x;
  long long total = (long long)rows * cols;
  if (i >= total) return;
  int r = (int)(i / cols);
  int c = (int)(i % cols);
  dst[(size_t)r * colsp + c] = (__bf16)src[i];
}

// ---- y = A(M x HIDP) @ W(N x HIDP)^T, bf16 in, f32 acc ----
// mode 0: scatter bf16 into q/k layout (bh, s, d) with DP padding
// mode 1: scatter bf16 into v-transposed layout (bh, d, s)
// mode 2: f32 row-major (M x HID) to d_out
__global__ void __launch_bounds__(256)
gemm_bf16_kernel(const __bf16* __restrict__ A, const __bf16* __restrict__ W,
                 void* __restrict__ out, int mode) {
  const int tid  = threadIdx.x;
  const int wid  = tid >> 5;
  const int lane = tid & 31;
  const int l16  = lane & 15;
  const int half = lane >> 4;
  const int wn = wid & 3, wm = wid >> 2;          // 2 M-waves x 4 N-waves
  const int nbase = blockIdx.x * 256 + wn * 64;   // wave: 32M x 64N tile
  const int mbase = blockIdx.y * 64  + wm * 32;

  v8f acc[2][4];
  for (int mi = 0; mi < 2; ++mi)
    for (int nt = 0; nt < 4; ++nt)
      for (int r = 0; r < 8; ++r) acc[mi][nt][r] = 0.0f;

  for (int kk = 0; kk < HIDP; kk += 32) {
    v16bf af[2];
    for (int mi = 0; mi < 2; ++mi) {
      const __bf16* ar = A + (size_t)(mbase + mi*16 + l16) * HIDP + kk;
      af[mi] = frag16(ar + half*16, ar + half*16 + 8);
    }
#pragma unroll
    for (int nt = 0; nt < 4; ++nt) {
      const __bf16* wr = W + (size_t)(nbase + nt*16 + l16) * HIDP + kk + half*16;
      v16bf bf = frag16(wr, wr + 8);
      acc[0][nt] = wmma_bf16(af[0], bf, acc[0][nt]);
      acc[1][nt] = wmma_bf16(af[1], bf, acc[1][nt]);
    }
  }

  for (int mi = 0; mi < 2; ++mi)
    for (int nt = 0; nt < 4; ++nt) {
      int n = nbase + nt*16 + l16;
      if (n >= HID) continue;
      for (int r = 0; r < 8; ++r) {
        int m = mbase + mi*16 + r + half*8;
        float v = acc[mi][nt][r];
        if (mode == 2) {
          ((float*)out)[(size_t)m * HID + n] = v;
        } else {
          int b = m >> 11, s = m & (Sq - 1);
          int h = n / Dd, d = n % Dd;
          size_t idx = (mode == 0)
              ? ((size_t)(b*Hh + h) * Sq + s) * DP + d
              : ((size_t)(b*Hh + h) * DP + d) * Sq + s;
          ((__bf16*)out)[idx] = (__bf16)v;
        }
      }
    }
}

// ---- flash attention: one (b,h) per blockIdx.x, 128 query rows per WG ----
__global__ void __launch_bounds__(256)
flash_attn_kernel(const __bf16* __restrict__ qb, const __bf16* __restrict__ kb,
                  const __bf16* __restrict__ vtb,
                  const float* __restrict__ alibi, const float* __restrict__ amask,
                  const int* __restrict__ lip, __bf16* __restrict__ ctx) {
  __shared__ alignas(16) __bf16 pst[8][16 * 32];   // per-wave P staging

  const int tid  = threadIdx.x;
  const int wid  = tid >> 5;
  const int lane = tid & 31;
  const int l16  = lane & 15;
  const int half = lane >> 4;
  const int bh = blockIdx.x;
  const int b  = bh / Hh, h = bh % Hh;
  const int m0 = blockIdx.y * 128 + wid * 16;      // query row base

  const float invL   = 1.0f / (float)(lip[0] + 1);
  const float scaleA = 10.0f * invL;               // sqrt(100)/(layer+1)

  v16bf qf[4];
  {
    const __bf16* qrow = qb + ((size_t)bh * Sq + m0 + l16) * DP;
#pragma unroll
    for (int dc = 0; dc < 4; ++dc)
      qf[dc] = frag16(qrow + dc*32 + half*16, qrow + dc*32 + half*16 + 8);
  }

  v8f oacc[8];
  for (int nc = 0; nc < 8; ++nc)
    for (int r = 0; r < 8; ++r) oacc[nc][r] = 0.0f;
  float mrun[8], lrun[8];
  for (int r = 0; r < 8; ++r) { mrun[r] = -3.0e38f; lrun[r] = 0.0f; }

  for (int kb0 = 0; kb0 < Sq; kb0 += 32) {
    float sv[2][8];
#pragma unroll
    for (int c = 0; c < 2; ++c) {
      v8f sc;
      for (int r = 0; r < 8; ++r) sc[r] = 0.0f;
      const __bf16* krow = kb + ((size_t)bh * Sq + kb0 + c*16 + l16) * DP;
#pragma unroll
      for (int dc = 0; dc < 4; ++dc) {
        v16bf bf = frag16(krow + dc*32 + half*16, krow + dc*32 + half*16 + 8);
        sc = wmma_bf16(qf[dc], bf, sc);
      }
      int t = kb0 + c*16 + l16;
      float add = alibi[(size_t)bh * Sq + t] * invL + amask[(size_t)b * Sq + t];
      for (int r = 0; r < 8; ++r) sv[c][r] = sc[r] * scaleA + add;
    }

    // online softmax (rows live across 16-lane halves)
    float corr[8];
#pragma unroll
    for (int r = 0; r < 8; ++r) {
      float mx = fmaxf(sv[0][r], sv[1][r]);
      for (int off = 1; off < 16; off <<= 1)
        mx = fmaxf(mx, __shfl_xor(mx, off, 32));
      float mn = fmaxf(mrun[r], mx);
      corr[r] = __expf(mrun[r] - mn);
      mrun[r] = mn;
      float p0 = __expf(sv[0][r] - mn);
      float p1 = __expf(sv[1][r] - mn);
      sv[0][r] = p0; sv[1][r] = p1;
      float rs = p0 + p1;
      for (int off = 1; off < 16; off <<= 1)
        rs += __shfl_xor(rs, off, 32);
      lrun[r] = lrun[r] * corr[r] + rs;
    }
    for (int nc = 0; nc < 8; ++nc)
      for (int r = 0; r < 8; ++r) oacc[nc][r] *= corr[r];

    // P (f32 C-layout) -> bf16 A-fragment via per-wave LDS tile
    for (int c = 0; c < 2; ++c)
      for (int r = 0; r < 8; ++r)
        pst[wid][(r + half*8) * 32 + c*16 + l16] = (__bf16)sv[c][r];
    __syncthreads();
    const __bf16* prow = &pst[wid][l16 * 32];
    v16bf pf = frag16(prow + half*16, prow + half*16 + 8);
    __syncthreads();

#pragma unroll
    for (int nc = 0; nc < 8; ++nc) {
      const __bf16* vrow = vtb + ((size_t)bh * DP + nc*16 + l16) * Sq + kb0 + half*16;
      v16bf vb = frag16(vrow, vrow + 8);
      oacc[nc] = wmma_bf16(pf, vb, oacc[nc]);
    }
  }

  // context store, reproducing BLOOM's raw (b*h,s,d)->(b,s,HID) reshape
  float invl[8];
  for (int r = 0; r < 8; ++r) invl[r] = 1.0f / lrun[r];
  for (int nc = 0; nc < 8; ++nc) {
    int d = nc*16 + l16;
    if (d >= Dd) continue;
    for (int r = 0; r < 8; ++r) {
      int s = m0 + r + half*8;
      long long f = ((long long)(h * Sq + s)) * Dd + d;
      int row = (int)(f / HID);
      int col = (int)(f % HID);
      ctx[((size_t)b * Sq + row) * HIDP + col] = (__bf16)(oacc[nc][r] * invl[r]);
    }
  }
}

extern "C" void kernel_launch(void* const* d_in, const int* in_sizes, int n_in,
                              void* d_out, int out_size, void* d_ws, size_t ws_size,
                              hipStream_t stream) {
  const float* x     = (const float*)d_in[0];
  const float* alibi = (const float*)d_in[1];
  const float* amask = (const float*)d_in[2];
  const float* wq    = (const float*)d_in[3];
  const float* wk    = (const float*)d_in[4];
  const float* wv    = (const float*)d_in[5];
  const float* wo    = (const float*)d_in[6];
  const int*   li    = (const int*)d_in[7];
  float*       out   = (float*)d_out;

  char* ws = (char*)d_ws;
  size_t off = 0;
  const size_t xb_sz  = (size_t)MROWS * HIDP * 2;
  const size_t w_sz   = (size_t)HIDP * HIDP * 2;
  const size_t qkv_sz = (size_t)BH * Sq * DP * 2;
  __bf16* xb   = (__bf16*)(ws + off); off += xb_sz;
  __bf16* wqb  = (__bf16*)(ws + off); off += w_sz;
  __bf16* wkb  = (__bf16*)(ws + off); off += w_sz;
  __bf16* wvb  = (__bf16*)(ws + off); off += w_sz;
  __bf16* wob  = (__bf16*)(ws + off); off += w_sz;
  __bf16* qbuf = (__bf16*)(ws + off); off += qkv_sz;
  __bf16* kbuf = (__bf16*)(ws + off); off += qkv_sz;
  __bf16* vtb  = (__bf16*)(ws + off); off += qkv_sz;
  __bf16* ctxb = (__bf16*)(ws + off); off += xb_sz;

  hipMemsetAsync(d_ws, 0, off, stream);

  long long nx = (long long)MROWS * HID;
  cvt_pad_kernel<<<(unsigned)((nx + 255) / 256), 256, 0, stream>>>(x, xb, MROWS, HID, HIDP);
  long long nw = (long long)HID * HID;
  unsigned gw = (unsigned)((nw + 255) / 256);
  cvt_pad_kernel<<<gw, 256, 0, stream>>>(wq, wqb, HID, HID, HIDP);
  cvt_pad_kernel<<<gw, 256, 0, stream>>>(wk, wkb, HID, HID, HIDP);
  cvt_pad_kernel<<<gw, 256, 0, stream>>>(wv, wvb, HID, HID, HIDP);
  cvt_pad_kernel<<<gw, 256, 0, stream>>>(wo, wob, HID, HID, HIDP);

  dim3 gg((HIDP + 255) / 256, MROWS / 64);   // 5 x 128
  gemm_bf16_kernel<<<gg, 256, 0, stream>>>(xb, wqb, qbuf, 0);
  gemm_bf16_kernel<<<gg, 256, 0, stream>>>(xb, wkb, kbuf, 0);
  gemm_bf16_kernel<<<gg, 256, 0, stream>>>(xb, wvb, vtb, 1);

  flash_attn_kernel<<<dim3(BH, Sq / 128), 256, 0, stream>>>(
      qbuf, kbuf, vtb, alibi, amask, li, ctxb);

  gemm_bf16_kernel<<<gg, 256, 0, stream>>>(ctxb, wob, (void*)out, 2);
}
